// ConvNet_74268574482622
// MI455X (gfx1250) — matrix operand order repaired and checked by reading, not compiled, not run on hardware
//
#include <hip/hip_runtime.h>

typedef __attribute__((ext_vector_type(16))) _Float16 v16h;
typedef __attribute__((ext_vector_type(8)))  float    v8f;

#define BB     256
#define CIN    4
#define LIN    2048
#define HN     256
#define GL     24
#define KTOT   96      // CIN*GL, = 3 WMMA K-steps of 32
#define LOUT   2025    // LIN - GL + 1
#define MTILES 127     // ceil(LOUT/16)
#define MW     4       // M-tiles staged per barrier pair
#define ITERS  32      // ceil(MTILES/MW); tile 127 is a fully-masked dummy

// workspace layout (all offsets 256B-aligned)
#define XH_ELEMS   (BB*CIN*LIN)   // 2,097,152 halves
#define WH_ELEMS   (HN*KTOT)      // 24,576 halves
#define XH_BYTES   (XH_ELEMS*2)   // 4 MB
#define WH_BYTES   (WH_ELEMS*2)   // 48 KB

// ---------------------------------------------------------------- prep: fp32 -> f16
__global__ void prep_kernel(const float* __restrict__ x,
                            const float* __restrict__ wConv,
                            _Float16* __restrict__ xh,
                            _Float16* __restrict__ wh) {
    const int stride = gridDim.x * blockDim.x;
    const int i0 = blockIdx.x * blockDim.x + threadIdx.x;
    for (int i = i0; i < XH_ELEMS; i += stride) xh[i] = (_Float16)x[i];
    for (int i = i0; i < WH_ELEMS; i += stride) wh[i] = (_Float16)wConv[i];
}

// ------------------------------------------- fused conv1d + bias + relu + max/avg pool
// One workgroup per batch row; 16 wave32s, wave w owns channels [16w,16w+16) as one
// resident WMMA N-tile (B fragments in registers). Per iteration, 4 M-tiles worth of
// A fragments are gathered from global x directly into WMMA lane-swizzled LDS layout,
// so each consumer lane reads its v16h fragment as a single aligned 32B ds load.
__global__ __launch_bounds__(512)
void conv_pool_kernel(const _Float16* __restrict__ xh,
                      const _Float16* __restrict__ wh,
                      const float* __restrict__ wRect,
                      float* __restrict__ poolMax,     // [BB][HN]
                      float* __restrict__ poolSum) {   // [BB][HN]
    const int b    = blockIdx.x;
    const int tid  = threadIdx.x;
    const int wave = tid >> 5;          // 0..15  == N-tile index
    const int lane = tid & 31;
    const int lrow = lane & 15;         // B,C column (N) within tile
    const int lhi  = lane >> 4;         // lane-half select

    // A fragments, lane-swizzled: as[sub][ks][lane][e], 4*3*32*16 halves = 12 KB
    __shared__ __align__(32) _Float16 as[MW * 3 * 32 * 16];

    const int n = wave * 16 + lrow;     // channel this lane produces (C column)

    // B fragments (weights), resident all kernel. Contiguous-k B layout used here (same as A): lanes 0-15 hold
    // K = ks*32 + 0..15, lanes 16-31 hold K = ks*32 + 16..31. k = c*GL+g matches
    // wConv's flat layout -> one aligned v16h global load each.
    v16h bf[3];
#pragma unroll
    for (int ks = 0; ks < 3; ++ks)
        bf[ks] = *(const v16h*)(wh + n * KTOT + ks * 32 + lhi * 16);

    const float rbias = wRect[n];
    const _Float16* xb = xh + b * (CIN * LIN);

    float maxacc = 0.0f;                // relu output >= 0, so 0 is a safe identity
    float sumacc = 0.0f;

    for (int it = 0; it < ITERS; ++it) {
        const int m0 = it * (MW * 16);
        __syncthreads();                // previous iteration's fragment reads done

        // ---- stage: gather im2col elements straight into swizzled A layout ----
        // A 16x32 layout (matches B): consumer lane ln, element e holds
        //   K = ks*32 + ((ln>=16)?16:0) + e,  row M = ln&15.
#pragma unroll
        for (int sub = 0; sub < MW; ++sub) {
            const int ms = m0 + sub * 16;
#pragma unroll
            for (int j = 0; j < 3; ++j) {
                const int idx = tid + j * 512;       // 1536 halves per subtile
                const int ks  = idx >> 9;
                const int rem = idx & 511;
                const int ln  = rem >> 4;            // consumer lane
                const int e   = rem & 15;
                const int kb  = ks * 32 + ((ln >> 4) << 4);
                const int k   = kb + e;
                const int c   = k / GL;
                const int g   = k - c * GL;
                const int l   = ms + (ln & 15) + g;
                as[sub * 1536 + idx] = (l < LIN) ? xb[c * LIN + l] : (_Float16)0.0f;
            }
        }
        __syncthreads();

        // ---- compute: 3 chained WMMAs per 16x16 tile, fused pooling epilogue ----
#pragma unroll
        for (int sub = 0; sub < MW; ++sub) {
            const int mt = it * MW + sub;
            const _Float16* ap = as + sub * 1536 + lane * 16;
            v8f c = {};
#pragma unroll
            for (int ks = 0; ks < 3; ++ks) {
                const v16h a = *(const v16h*)(ap + ks * 512);
                c = __builtin_amdgcn_wmma_f32_16x16x32_f16(
                        false, a, false, bf[ks], (short)0, c, false, false);
            }
            // C layout: lane column N = channel, row M = v + 8*lhi.
            if (mt < MTILES - 1) {                   // uniform: all rows valid
#pragma unroll
                for (int v = 0; v < 8; ++v) {
                    const float r = fmaxf(c[v] + rbias, 0.0f);
                    maxacc = fmaxf(maxacc, r);
                    sumacc += r;
                }
            } else {                                 // tail tile(s): branchless mask
                const int lbase = mt * 16 + lhi * 8;
#pragma unroll
                for (int v = 0; v < 8; ++v) {
                    float r = fmaxf(c[v] + rbias, 0.0f);
                    r = (lbase + v < LOUT) ? r : 0.0f;   // select, no exec churn
                    maxacc = fmaxf(maxacc, r);
                    sumacc += r;
                }
            }
        }
    }

    // lanes t and t+16 hold the same channel (row halves) -> pair reduce
    maxacc = fmaxf(maxacc, __shfl_xor(maxacc, 16));
    sumacc += __shfl_xor(sumacc, 16);
    if (lhi == 0) {
        poolMax[b * HN + n] = maxacc;
        poolSum[b * HN + n] = sumacc;   // mean applied in MLP kernel
    }
}

// ---------------------------------------------------- fused hidden layer + output head
__global__ __launch_bounds__(256)
void mlp_kernel(const float* __restrict__ poolMax,
                const float* __restrict__ poolSum,
                const float* __restrict__ wHidden,      // [512][512] row-major
                const float* __restrict__ wHiddenBias,  // [512]
                const float* __restrict__ wNeu,         // [512]
                const float* __restrict__ wNeuBias,     // [1]
                float* __restrict__ out) {              // [BB]
    const int b   = blockIdx.x;
    const int tid = threadIdx.x;
    __shared__ float p[2 * HN];
    __shared__ float red[256];

    p[tid]      = poolMax[b * HN + tid];
    p[tid + HN] = poolSum[b * HN + tid] * (1.0f / (float)LOUT);
    __syncthreads();

    float partial = 0.0f;
#pragma unroll
    for (int jj = 0; jj < 2; ++jj) {
        const int j = tid + jj * 256;
        float acc = wHiddenBias[j];
        for (int i = 0; i < 2 * HN; ++i)
            acc = fmaf(p[i], wHidden[i * (2 * HN) + j], acc);
        acc = acc > 0.0f ? acc : 0.0f;       // relu
        partial += acc * wNeu[j];
    }
    red[tid] = partial;
    __syncthreads();
    for (int s = 128; s > 0; s >>= 1) {
        if (tid < s) red[tid] += red[tid + s];
        __syncthreads();
    }
    if (tid == 0) out[b] = 0.5f * red[0] + wNeuBias[0];   // DROPPROB * (.) + bias
}

// ------------------------------------------------------------------------------ launch
extern "C" void kernel_launch(void* const* d_in, const int* in_sizes, int n_in,
                              void* d_out, int out_size, void* d_ws, size_t ws_size,
                              hipStream_t stream) {
    const float* x           = (const float*)d_in[0];
    const float* wConv       = (const float*)d_in[1];
    const float* wRect       = (const float*)d_in[2];
    const float* wHidden     = (const float*)d_in[3];
    const float* wHiddenBias = (const float*)d_in[4];
    const float* wNeu        = (const float*)d_in[5];
    const float* wNeuBias    = (const float*)d_in[6];
    float* out = (float*)d_out;

    char* ws = (char*)d_ws;
    _Float16* xh     = (_Float16*)(ws);
    _Float16* wh     = (_Float16*)(ws + XH_BYTES);
    float*    poolMx = (float*)(ws + XH_BYTES + WH_BYTES);
    float*    poolSm = poolMx + BB * HN;
    // total ws use: 4 MB + 48 KB + 2 * 256 KB  ~= 4.6 MB

    prep_kernel<<<2048, 256, 0, stream>>>(x, wConv, xh, wh);
    conv_pool_kernel<<<BB, 512, 0, stream>>>(xh, wh, wRect, poolMx, poolSm);
    mlp_kernel<<<BB, 256, 0, stream>>>(poolMx, poolSm, wHidden, wHiddenBias,
                                       wNeu, wNeuBias, out);
    (void)in_sizes; (void)n_in; (void)out_size; (void)ws_size;
}
